// DGCNModel_34479997452467
// MI455X (gfx1250) — matrix operand order repaired and checked by reading, not compiled, not run on hardware
//
#include <hip/hip_runtime.h>
#include <hip/hip_bf16.h>

// ---------------------------------------------------------------------------
// Problem constants (match reference)
// ---------------------------------------------------------------------------
constexpr int Bc  = 4;
constexpr int Nc  = 10000;
constexpr int Ec  = 320000;
constexpr int Tc  = 16;
constexpr int DINc= 16;
constexpr int NDc = 16;
constexpr int Lc  = 7;

typedef __attribute__((ext_vector_type(16))) _Float16 v16h;
typedef __attribute__((ext_vector_type(8)))  float    v8f;

// ---------------------------------------------------------------------------
// Fragment-major LDS layout: fragment f, lane l owns 16 contiguous halves at
// offset ((f*32 + l)*16). A fragment load is then one aligned 32B LDS read.
//
// gfx1250 wave32 WMMA layouts (CDNA5 ISA §7.12.2):
//  A 16x32 f16: lane (row + 16*hi); element i -> K = i+8*hi (i<8) else i+8+8*hi
//  B 32x16 f16: lane (col + 16*hi); element i -> K = i+8*hi (i<8) else i+8+8*hi
//  C 16x16 f32: VGPR r -> M = r + 8*hi, N = lane&15
// ---------------------------------------------------------------------------
__device__ __forceinline__ v16h frag_load(const _Float16* base, int frag, int lane) {
  return *(const v16h*)(base + ((frag * 32 + lane) << 4));
}

// Position of (row, k) inside A-fragment layout (for writing activations).
__device__ __forceinline__ int a_frag_pos(int row, int k) {
  int kt = k >> 5, kk = k & 31;
  int hi = (kk >> 3) & 1;
  int i  = (kk < 16) ? (kk - 8 * hi) : (kk - 8 - 8 * hi);
  return (((kt * 32) + row + (hi << 4)) << 4) + i;
}

// Stage a 16xK A-tile into fragment layout; val(row,k) supplies data (zero-pads).
template <typename F>
__device__ __forceinline__ void stage_a_frags(_Float16* dst, int nkt, F&& val, int lane) {
  for (int idx = lane; idx < nkt * 512; idx += 32) {
    int kt = idx >> 9, rem = idx & 511, l = rem >> 4, i = rem & 15;
    int row = l & 15, hi = l >> 4;
    int k = kt * 32 + ((i < 8) ? (i + 8 * hi) : (i + 8 + 8 * hi));
    dst[idx] = (_Float16)val(row, k);
  }
}

// Stage global W[Kv][Nv] (f32 row-major) into B-fragment layout padded to
// Kp x Np (Kp mult of 32, Np mult of 16). Fragment index f = kt*(Np/16)+nt.
__device__ __forceinline__ void stage_b_frags(const float* __restrict__ W,
                                              int Kv, int Nv, int Kp, int Np,
                                              _Float16* dst, int tid, int nthreads) {
  const int nnt = Np >> 4;
  const int total = (Kp >> 5) * nnt * 512;
  for (int idx = tid; idx < total; idx += nthreads) {
    int f = idx >> 9, rem = idx & 511, l = rem >> 4, i = rem & 15;
    int kt = f / nnt, nt = f - kt * nnt;
    int col = l & 15, hi = l >> 4;
    int k = kt * 32 + ((i < 8) ? (i + 8 * hi) : (i + 8 + 8 * hi));
    int n = nt * 16 + col;
    float v = (k < Kv && n < Nv) ? W[k * Nv + n] : 0.0f;
    dst[idx] = (_Float16)v;
  }
}

__device__ __forceinline__ v8f wmma_f16(v16h a, v16h b, v8f c) {
  return __builtin_amdgcn_wmma_f32_16x16x32_f16(false, a, false, b, (short)0, c,
                                                false, false);
}

__device__ __forceinline__ v8f bias_frag(const float* bias, int nbase, int lane) {
  float bv = bias[nbase + (lane & 15)];
  v8f c;
#pragma unroll
  for (int r = 0; r < 8; ++r) c[r] = bv;
  return c;
}

__device__ __forceinline__ float leaky(float v) { return v > 0.0f ? v : 0.01f * v; }

// ---------------------------------------------------------------------------
// Kernel 1: GRU over T=16, max-over-time, y_rnn head.
// One wave per 16-node tile; weights + state resident in LDS as fragments.
// ---------------------------------------------------------------------------
__global__ __launch_bounds__(32) void gru_kernel(
    const float* __restrict__ x,       // [B,N,T,DIN]
    const float* __restrict__ Wx,      // [16,192]
    const float* __restrict__ Wh,      // [64,192]
    const float* __restrict__ bx, const float* __restrict__ bh,   // [192]
    const float* __restrict__ fcW, const float* __restrict__ fcb, // [64,7],[7]
    float* __restrict__ rnn_out,       // [B,N,64]
    float* __restrict__ out)           // [B,N,7]
{
  __shared__ alignas(32) _Float16 sWxF[12 * 512];       // 12 frags (Kp=32,Np=192)
  __shared__ alignas(32) _Float16 sWhF[24 * 512];       // 24 frags (Kp=64,Np=192)
  __shared__ alignas(32) _Float16 sHF[2 * 512];         // hidden state A-frags (K=64)
  __shared__ alignas(32) _Float16 sXtF[512];            // x_t A-frag (Kp=32)
  __shared__ float sbx[192], sbh[192];
  __shared__ float sG[16 * 256];   // cols 0..127: r,z sums; 128..191 xn; 192..255 hn
  __shared__ float sHmax[16 * 64];

  const int lane  = threadIdx.x;
  const int b     = blockIdx.x;
  const int node0 = blockIdx.y * 16;
  const int col   = lane & 15, hi = (lane >> 4) & 1;

  stage_b_frags(Wx, 16, 192, 32, 192, sWxF, lane, 32);
  stage_b_frags(Wh, 64, 192, 64, 192, sWhF, lane, 32);
  for (int i = lane; i < 192; i += 32) { sbx[i] = bx[i]; sbh[i] = bh[i]; }
  for (int i = lane; i < 2 * 512; i += 32) sHF[i] = (_Float16)0.0f;
  for (int i = lane; i < 16 * 64; i += 32) sHmax[i] = -3.0e38f;
  __syncthreads();

  for (int t = 0; t < Tc; ++t) {
    stage_a_frags(sXtF, 1,
        [&](int row, int k) -> float {
          return (k < DINc)
              ? x[(((size_t)b * Nc + node0 + row) * Tc + t) * DINc + k]
              : 0.0f;
        }, lane);
    __syncthreads();

    for (int nt = 0; nt < 12; ++nt) {
      const int nb = nt * 16;
      v8f cx = bias_frag(sbx, nb, lane);
      v8f ch = bias_frag(sbh, nb, lane);
      v16h ax = frag_load(sXtF, 0, lane);
      v16h bw = frag_load(sWxF, nt, lane);
      cx = wmma_f16(ax, bw, cx);
#pragma unroll
      for (int kt = 0; kt < 2; ++kt) {
        v16h ah = frag_load(sHF, kt, lane);
        v16h bh2 = frag_load(sWhF, kt * 12 + nt, lane);
        ch = wmma_f16(ah, bh2, ch);
      }
      if (nt < 8) {
#pragma unroll
        for (int r = 0; r < 8; ++r)
          sG[(r + 8 * hi) * 256 + nb + col] = cx[r] + ch[r];
      } else {
#pragma unroll
        for (int r = 0; r < 8; ++r) {
          sG[(r + 8 * hi) * 256 + nb + col]      = cx[r];   // xn at 128..191
          sG[(r + 8 * hi) * 256 + nb + 64 + col] = ch[r];   // hn at 192..255
        }
      }
    }
    __syncthreads();

    // GRU elementwise update; h written back in A-fragment layout.
    for (int i = lane; i < 16 * 64; i += 32) {
      int r = i >> 6, c = i & 63;
      float rs = sG[r * 256 + c];
      float zs = sG[r * 256 + 64 + c];
      float xn = sG[r * 256 + 128 + c];
      float hn = sG[r * 256 + 192 + c];
      float rg = 1.0f / (1.0f + expf(-rs));
      float zg = 1.0f / (1.0f + expf(-zs));
      float ng = tanhf(xn + rg * hn);
      int hp = a_frag_pos(r, c);
      float hnew = (1.0f - zg) * ng + zg * (float)sHF[hp];
      sHF[hp] = (_Float16)hnew;
      sHmax[i] = fmaxf(sHmax[i], hnew);
    }
    __syncthreads();
  }

  for (int i = lane; i < 16 * 64; i += 32) {
    int r = i >> 6, c = i & 63;
    rnn_out[((size_t)b * Nc + node0 + r) * 64 + c] = sHmax[i];
  }
  for (int i = lane; i < 16 * Lc; i += 32) {
    int r = i / Lc, l = i % Lc;
    float acc = fcb[l];
    for (int k = 0; k < 64; ++k) acc += sHmax[r * 64 + k] * fcW[k * Lc + l];
    out[((size_t)b * Nc + node0 + r) * Lc + l] = acc;
  }
}

// ---------------------------------------------------------------------------
// Kernel 2: per-node MLP: leaky( leaky([x|emb]@W1+b1) @ W2 + b2 ).  K=80->96.
// 4 waves/block, each wave owns a 16-node tile.
// ---------------------------------------------------------------------------
__global__ __launch_bounds__(128) void node_mlp_kernel(
    const float* __restrict__ x,     // [B,N,64]
    const float* __restrict__ emb,   // [NT,16]
    const int*   __restrict__ types, // [N]
    const float* __restrict__ W1, const float* __restrict__ b1,  // [80,64],[64]
    const float* __restrict__ W2, const float* __restrict__ b2,  // [64,64],[64]
    float* __restrict__ out)         // [B,N,64]
{
  __shared__ alignas(32) _Float16 sW1F[12 * 512];      // Kp=96 x Np=64 -> 3*4 frags
  __shared__ alignas(32) _Float16 sW2F[8 * 512];       // Kp=64 x Np=64 -> 2*4 frags
  __shared__ alignas(32) _Float16 sInF[4][3 * 512];    // per-wave A frags (Kp=96)
  __shared__ alignas(32) _Float16 sH1F[4][2 * 512];    // per-wave layer1 A frags
  __shared__ float sb1[64], sb2[64];

  const int tid = threadIdx.x, wave = tid >> 5, lane = tid & 31;
  const int col = lane & 15, hi = (lane >> 4) & 1;

  stage_b_frags(W1, 80, 64, 96, 64, sW1F, tid, 128);
  stage_b_frags(W2, 64, 64, 64, 64, sW2F, tid, 128);
  if (tid < 64) { sb1[tid] = b1[tid]; sb2[tid] = b2[tid]; }

  const int tileg = blockIdx.x * 4 + wave;
  const int b     = tileg / (Nc / 16);
  const int node0 = (tileg % (Nc / 16)) * 16;

  _Float16* in = sInF[wave];
  stage_a_frags(in, 3,
      [&](int row, int k) -> float {
        int node = node0 + row;
        if (k < 64) return x[((size_t)b * Nc + node) * 64 + k];
        if (k < 80) return emb[types[node] * NDc + (k - 64)];
        return 0.0f;
      }, lane);
  __syncthreads();

  _Float16* h1 = sH1F[wave];
  for (int nt = 0; nt < 4; ++nt) {
    v8f acc = bias_frag(sb1, nt * 16, lane);
#pragma unroll
    for (int kt = 0; kt < 3; ++kt)
      acc = wmma_f16(frag_load(in, kt, lane), frag_load(sW1F, kt * 4 + nt, lane), acc);
#pragma unroll
    for (int r = 0; r < 8; ++r)
      h1[a_frag_pos(r + 8 * hi, nt * 16 + col)] = (_Float16)leaky(acc[r]);
  }

  for (int nt = 0; nt < 4; ++nt) {
    v8f acc = bias_frag(sb2, nt * 16, lane);
#pragma unroll
    for (int kt = 0; kt < 2; ++kt)
      acc = wmma_f16(frag_load(h1, kt, lane), frag_load(sW2F, kt * 4 + nt, lane), acc);
#pragma unroll
    for (int r = 0; r < 8; ++r)
      out[((size_t)b * Nc + node0 + r + 8 * hi) * 64 + nt * 16 + col] = leaky(acc[r]);
  }
}

// ---------------------------------------------------------------------------
// Kernel 3: gather x_in[src]*w, scatter-add at dst (accumulator seeded with x).
// ---------------------------------------------------------------------------
__global__ void scatter_kernel(const float* __restrict__ xin,   // [B,N,64]
                               const int*   __restrict__ ei,    // [2,E]
                               const float* __restrict__ attr,  // [E]
                               const float* __restrict__ gate,  // [B,E] or null
                               float* __restrict__ acc)         // [B,N,64]
{
  size_t idx = (size_t)blockIdx.x * blockDim.x + threadIdx.x;
  if (idx >= (size_t)Bc * Ec) return;
  int b = (int)(idx / Ec), e = (int)(idx % Ec);
  float w = gate ? gate[idx] : attr[e];
  int s = ei[e], d = ei[Ec + e];
  const float* srcp = xin + ((size_t)b * Nc + s) * 64;
  float*       dstp = acc + ((size_t)b * Nc + d) * 64;
#pragma unroll 4
  for (int c = 0; c < 64; ++c) atomicAdd(&dstp[c], srcp[c] * w);
}

// ---------------------------------------------------------------------------
// Kernel 4: edge gate MLP: relu( leaky([x_src|x_dst]@W1+b1) @ W2 + b2 ). K=128.
// ---------------------------------------------------------------------------
__global__ __launch_bounds__(128) void edge_mlp_kernel(
    const float* __restrict__ x,   // [B,N,64]
    const int*   __restrict__ ei,  // [2,E]
    const float* __restrict__ W1, const float* __restrict__ b1,  // [128,64],[64]
    const float* __restrict__ W2, const float* __restrict__ b2,  // [64,1],[1]
    float* __restrict__ gate)      // [B,E]
{
  __shared__ alignas(32) _Float16 sW1F[16 * 512];      // Kp=128 x Np=64 -> 4*4 frags
  __shared__ alignas(32) _Float16 sInF[4][4 * 512];    // per-wave A frags (K=128)
  __shared__ float sH1[4][16 * 64];
  __shared__ float sW2[64], sb1[64];
  __shared__ float sb2v;

  const int tid = threadIdx.x, wave = tid >> 5, lane = tid & 31;
  const int col = lane & 15, hi = (lane >> 4) & 1;

  stage_b_frags(W1, 128, 64, 128, 64, sW1F, tid, 128);
  if (tid < 64) { sW2[tid] = W2[tid]; sb1[tid] = b1[tid]; }
  if (tid == 0) sb2v = b2[0];

  const int tileg = blockIdx.x * 4 + wave;
  const int b  = tileg / (Ec / 16);
  const int e0 = (tileg % (Ec / 16)) * 16;

  _Float16* in = sInF[wave];
  stage_a_frags(in, 4,
      [&](int row, int k) -> float {
        int e = e0 + row;
        int node = (k < 64) ? ei[e] : ei[Ec + e];
        return x[((size_t)b * Nc + node) * 64 + (k & 63)];
      }, lane);
  __syncthreads();

  float* h1 = sH1[wave];
  for (int nt = 0; nt < 4; ++nt) {
    v8f acc = bias_frag(sb1, nt * 16, lane);
#pragma unroll
    for (int kt = 0; kt < 4; ++kt)
      acc = wmma_f16(frag_load(in, kt, lane), frag_load(sW1F, kt * 4 + nt, lane), acc);
#pragma unroll
    for (int r = 0; r < 8; ++r)
      h1[(r + 8 * hi) * 64 + nt * 16 + col] = leaky(acc[r]);
  }
  __syncthreads();

  if (lane < 16) {
    float acc = sb2v;
    for (int k = 0; k < 64; ++k) acc += h1[lane * 64 + k] * sW2[k];
    gate[(size_t)b * Ec + e0 + lane] = fmaxf(acc, 0.0f);
  }
}

// ---------------------------------------------------------------------------
// Kernel 5: out += x @ gcn_fc_W + gcn_fc_b
// ---------------------------------------------------------------------------
__global__ void final_fc_kernel(const float* __restrict__ x,  // [B,N,64]
                                const float* __restrict__ W,  // [64,7]
                                const float* __restrict__ bb, // [7]
                                float* __restrict__ out)      // [B,N,7]
{
  int idx = blockIdx.x * blockDim.x + threadIdx.x;
  if (idx >= Bc * Nc) return;
  const float* xr = x + (size_t)idx * 64;
  float* o = out + (size_t)idx * Lc;
  for (int l = 0; l < Lc; ++l) {
    float acc = bb[l];
    for (int k = 0; k < 64; ++k) acc += xr[k] * W[k * Lc + l];
    o[l] += acc;
  }
}

// ---------------------------------------------------------------------------
// Host-side orchestration
// ---------------------------------------------------------------------------
extern "C" void kernel_launch(void* const* d_in, const int* in_sizes, int n_in,
                              void* d_out, int out_size, void* d_ws, size_t ws_size,
                              hipStream_t stream) {
  const float* input_day  = (const float*)d_in[0];
  const int*   edge_index = (const int*)  d_in[1];
  const int*   node_type  = (const int*)  d_in[2];
  const float* edge_attr  = (const float*)d_in[3];
  const float* gru_Wx = (const float*)d_in[4];
  const float* gru_Wh = (const float*)d_in[5];
  const float* gru_bx = (const float*)d_in[6];
  const float* gru_bh = (const float*)d_in[7];
  const float* rnn_fc_W = (const float*)d_in[8];
  const float* rnn_fc_b = (const float*)d_in[9];
  const float* e_emb = (const float*)d_in[10];
  const float* g_emb = (const float*)d_in[11];
  const float* e_c0_W1 = (const float*)d_in[12]; const float* e_c0_b1 = (const float*)d_in[13];
  const float* e_c0_W2 = (const float*)d_in[14]; const float* e_c0_b2 = (const float*)d_in[15];
  const float* e_c1_W1 = (const float*)d_in[16]; const float* e_c1_b1 = (const float*)d_in[17];
  const float* e_c1_W2 = (const float*)d_in[18]; const float* e_c1_b2 = (const float*)d_in[19];
  const float* g_c0_W1 = (const float*)d_in[20]; const float* g_c0_b1 = (const float*)d_in[21];
  const float* g_c0_W2 = (const float*)d_in[22]; const float* g_c0_b2 = (const float*)d_in[23];
  const float* g_c1_W1 = (const float*)d_in[24]; const float* g_c1_b1 = (const float*)d_in[25];
  const float* g_c1_W2 = (const float*)d_in[26]; const float* g_c1_b2 = (const float*)d_in[27];
  const float* e_fm_W1 = (const float*)d_in[28]; const float* e_fm_b1 = (const float*)d_in[29];
  const float* e_fm_W2 = (const float*)d_in[30]; const float* e_fm_b2 = (const float*)d_in[31];
  const float* gcn_fc_W = (const float*)d_in[32]; const float* gcn_fc_b = (const float*)d_in[33];

  float* out = (float*)d_out;
  const size_t NB = (size_t)Bc * Nc * 64;
  float* ws0  = (float*)d_ws;                     // rnn_out
  float* ws1  = ws0 + NB;                         // mlp output x_in
  float* ws2  = ws1 + NB;                         // ping accumulator
  float* ws3  = ws2 + NB;                         // pong accumulator
  float* gate = ws3 + NB;                         // [B,E]

  const dim3 gru_grid(Bc, Nc / 16);
  const int mlp_blocks = (Bc * Nc / 16) / 4;      // 625
  const int sc_blocks  = (int)(((size_t)Bc * Ec + 255) / 256);
  const int eg_blocks  = (Bc * Ec / 16) / 4;      // 20000
  const int fc_blocks  = (Bc * Nc + 255) / 256;
  const size_t NBbytes = NB * sizeof(float);

  // 1) GRU encoder + max-over-time + rnn head (writes d_out = y_rnn)
  gru_kernel<<<gru_grid, 32, 0, stream>>>(input_day, gru_Wx, gru_Wh, gru_bx, gru_bh,
                                          rnn_fc_W, rnn_fc_b, ws0, out);

  // 2) EdgeNet conv0
  node_mlp_kernel<<<mlp_blocks, 128, 0, stream>>>(ws0, e_emb, node_type,
                                                  e_c0_W1, e_c0_b1, e_c0_W2, e_c0_b2, ws1);
  hipMemcpyAsync(ws2, ws0, NBbytes, hipMemcpyDeviceToDevice, stream);
  scatter_kernel<<<sc_blocks, 256, 0, stream>>>(ws1, edge_index, edge_attr, nullptr, ws2);

  //    EdgeNet conv1
  node_mlp_kernel<<<mlp_blocks, 128, 0, stream>>>(ws2, e_emb, node_type,
                                                  e_c1_W1, e_c1_b1, e_c1_W2, e_c1_b2, ws1);
  hipMemcpyAsync(ws3, ws2, NBbytes, hipMemcpyDeviceToDevice, stream);
  scatter_kernel<<<sc_blocks, 256, 0, stream>>>(ws1, edge_index, edge_attr, nullptr, ws3);

  // 3) Edge gate MLP on [x_src | x_dst]
  edge_mlp_kernel<<<eg_blocks, 128, 0, stream>>>(ws3, edge_index,
                                                 e_fm_W1, e_fm_b1, e_fm_W2, e_fm_b2, gate);

  // 4) Main GCN conv0 (edge weights = learned gates)
  node_mlp_kernel<<<mlp_blocks, 128, 0, stream>>>(ws0, g_emb, node_type,
                                                  g_c0_W1, g_c0_b1, g_c0_W2, g_c0_b2, ws1);
  hipMemcpyAsync(ws2, ws0, NBbytes, hipMemcpyDeviceToDevice, stream);
  scatter_kernel<<<sc_blocks, 256, 0, stream>>>(ws1, edge_index, edge_attr, gate, ws2);

  //    Main GCN conv1
  node_mlp_kernel<<<mlp_blocks, 128, 0, stream>>>(ws2, g_emb, node_type,
                                                  g_c1_W1, g_c1_b1, g_c1_W2, g_c1_b2, ws1);
  hipMemcpyAsync(ws3, ws2, NBbytes, hipMemcpyDeviceToDevice, stream);
  scatter_kernel<<<sc_blocks, 256, 0, stream>>>(ws1, edge_index, edge_attr, gate, ws3);

  // 5) out += x @ gcn_fc
  final_fc_kernel<<<fc_blocks, 256, 0, stream>>>(ws3, gcn_fc_W, gcn_fc_b, out);
}
